// QEffGrok1MoeBlock_52269751992572
// MI455X (gfx1250) — matrix operand order repaired and checked by reading, not compiled, not run on hardware
//
#include <hip/hip_runtime.h>
#include <hip/hip_bf16.h>
#include <math.h>

// Problem dims (B=2,S=2048,H=2048,I=8192,E=8)
#define T_TOK 4096
#define H_DIM 2048
#define I_DIM 8192
#define N_EXP 8
#define PAIRS 8192

// GEMM tile params
#define TM 128   // token rows per block (8 waves x 16)
#define TN 64    // output cols per block (4 wmma n-tiles)
#define TK 32    // K step (bf16 wmma K)
#define KP 40    // padded LDS K-stride in elements (80B -> conflict-free rows)

typedef __attribute__((ext_vector_type(16))) __bf16 v16bf;
typedef __attribute__((ext_vector_type(8)))  __bf16 v8bf;
typedef __attribute__((ext_vector_type(8)))  float  v8f;

union AF { v16bf v; uint4 u[2]; };

// packed fp32 -> bf16 (RTE) via vector convert: lets the backend pick
// native packed cvt instructions instead of a manual shift/add chain.
__device__ __forceinline__ uint4 pack8(float4 f0, float4 f1) {
  v8f f = {f0.x, f0.y, f0.z, f0.w, f1.x, f1.y, f1.z, f1.w};
  union { v8bf b; uint4 u; } c;
  c.b = __builtin_convertvector(f, v8bf);
  return c.u;
}

__device__ __forceinline__ unsigned short bf16bits(float f) {
  union { __bf16 b; unsigned short s; } c;
  c.b = (__bf16)f;
  return c.s;
}

// Async global->LDS 16B copies (CDNA5, ASYNCcnt-tracked). INST_OFFSET is
// applied to both the global and LDS addresses.
#define ASYNC_LD16(ldsOff, gaddr)                                            \
  asm volatile("global_load_async_to_lds_b128 %0, %1, off"                   \
               :: "v"(ldsOff), "v"(gaddr) : "memory")
#define ASYNC_LD16_O16(ldsOff, gaddr)                                        \
  asm volatile("global_load_async_to_lds_b128 %0, %1, off offset:16"         \
               :: "v"(ldsOff), "v"(gaddr) : "memory")
#define WAIT_ASYNC0() asm volatile("s_wait_asynccnt 0x0" ::: "memory")

// ---------------- x fp32 -> bf16 pre-conversion -----------------------------
__global__ __launch_bounds__(256) void k_cvtx(const float* __restrict__ x,
                                              unsigned short* __restrict__ xbf) {
  size_t i = ((size_t)blockIdx.x * 256 + threadIdx.x) * 8;
  const float4* p = (const float4*)(x + i);
  *(uint4*)(xbf + i) = pack8(p[0], p[1]);
}

// ---------------- Router: logits = x @ gate_w^T -----------------------------
__global__ __launch_bounds__(256) void k_router(const float* __restrict__ x,
                                                const float* __restrict__ gw,
                                                float* __restrict__ logits) {
  int wave = threadIdx.x >> 5, lane = threadIdx.x & 31;
  int t = blockIdx.x * 8 + wave;               // one wave per token
  float acc[N_EXP];
#pragma unroll
  for (int e = 0; e < N_EXP; ++e) acc[e] = 0.f;
  const float* xr = x + (size_t)t * H_DIM;
  for (int h = lane; h < H_DIM; h += 32) {
    float xv = xr[h];
#pragma unroll
    for (int e = 0; e < N_EXP; ++e) acc[e] += xv * gw[e * H_DIM + h];
  }
#pragma unroll
  for (int e = 0; e < N_EXP; ++e) {
#pragma unroll
    for (int off = 16; off >= 1; off >>= 1)
      acc[e] += __shfl_xor(acc[e], off, 32);
  }
  if (lane == 0) {
#pragma unroll
    for (int e = 0; e < N_EXP; ++e) logits[t * N_EXP + e] = acc[e];
  }
}

// ---------------- Softmax + top-2 ------------------------------------------
__global__ __launch_bounds__(256) void k_top2(const float* __restrict__ logits,
                                              int* __restrict__ i1, int* __restrict__ i2,
                                              float* __restrict__ w1, float* __restrict__ w2) {
  int t = blockIdx.x * blockDim.x + threadIdx.x;
  if (t >= T_TOK) return;
  float l[N_EXP];
  float m = -3.0e38f;
#pragma unroll
  for (int e = 0; e < N_EXP; ++e) { l[e] = logits[t * N_EXP + e]; m = fmaxf(m, l[e]); }
  float s = 0.f;
#pragma unroll
  for (int e = 0; e < N_EXP; ++e) { l[e] = __expf(l[e] - m); s += l[e]; }
  float inv = 1.f / s;
  int a = 0; float pa = l[0];
#pragma unroll
  for (int e = 1; e < N_EXP; ++e) if (l[e] > pa) { pa = l[e]; a = e; }
  int b = (a == 0) ? 1 : 0; float pb = l[b];
#pragma unroll
  for (int e = 0; e < N_EXP; ++e) if (e != a && l[e] > pb) { pb = l[e]; b = e; }
  i1[t] = a; i2[t] = b; w1[t] = pa * inv; w2[t] = pb * inv;
}

// ---------------- Deterministic routing build (1 block, wave-per-expert) ----
__global__ __launch_bounds__(256) void k_build(const int* __restrict__ i1, const int* __restrict__ i2,
                                               const float* __restrict__ w1, const float* __restrict__ w2,
                                               int* __restrict__ eoff,
                                               int* __restrict__ perm_tok, float* __restrict__ perm_w) {
  __shared__ int scnt[N_EXP];
  __shared__ int soff[N_EXP + 1];
  int wave = threadIdx.x >> 5, lane = threadIdx.x & 31;
  int e = wave;
  int cnt = 0;
  for (int t0 = 0; t0 < T_TOK; t0 += 32) {
    int t = t0 + lane;
    int m = (i1[t] == e || i2[t] == e) ? 1 : 0;
#pragma unroll
    for (int off = 16; off >= 1; off >>= 1) m += __shfl_xor(m, off, 32);
    cnt += m;
  }
  if (lane == 0) scnt[e] = cnt;
  __syncthreads();
  if (threadIdx.x == 0) {
    int o = 0;
#pragma unroll
    for (int k = 0; k < N_EXP; ++k) { soff[k] = o; eoff[k] = o; o += scnt[k]; }
    soff[N_EXP] = o; eoff[N_EXP] = o;
  }
  __syncthreads();
  int pos = soff[e];
  for (int t0 = 0; t0 < T_TOK; t0 += 32) {
    int t = t0 + lane;
    int m1 = (i1[t] == e);
    int m2 = (i2[t] == e);
    int m = (m1 || m2) ? 1 : 0;
    int scan = m;
#pragma unroll
    for (int d = 1; d < 32; d <<= 1) {
      int v = __shfl_up(scan, d, 32);
      if (lane >= d) scan += v;
    }
    if (m) {
      int idx = pos + scan - 1;
      perm_tok[idx] = t;
      perm_w[idx] = m1 ? w1[t] : w2[t];
    }
    pos += __shfl(scan, 31, 32);
  }
}

// ---------------- GEMM1: up = gelu(Xe@Wg^T) * (Xe@Wu^T) --------------------
__global__ __launch_bounds__(256) void k_up(const unsigned short* __restrict__ xbf,
                                            const float* __restrict__ wg,
                                            const float* __restrict__ wu,
                                            const int* __restrict__ eoff,
                                            const int* __restrict__ perm_tok,
                                            unsigned short* __restrict__ upbuf) {
  int e = blockIdx.y;
  int base = eoff[e], count = eoff[e + 1] - base;
  int mb = blockIdx.x;
  if (mb * TM >= count) return;
  int rowsLeft = count - mb * TM; if (rowsLeft > TM) rowsLeft = TM;
  int n0 = blockIdx.z * TN;

  __shared__ unsigned short lA[TM * KP];   // tokens  [M][K] bf16
  __shared__ unsigned short lBg[TN * KP];  // w_gate  [N][K] bf16
  __shared__ unsigned short lBu[TN * KP];  // w_up    [N][K] bf16

  int tid = threadIdx.x;
  int wave = tid >> 5, lane = tid & 31;

  // A (tokens): async bf16 copy, 32B per thread. Tail rows use token 0; their
  // GEMM rows are independent and never stored (epilogue guard).
  int arow = tid >> 1, acol = (tid & 1) * 16;
  int tokA = (arow < rowsLeft) ? perm_tok[base + mb * TM + arow] : 0;
  unsigned long long gaA =
      (unsigned long long)(size_t)(xbf + (size_t)tokA * H_DIM + acol);
  unsigned ldsA = (unsigned)(size_t)&lA[arow * KP + acol];

  // B (weights): fp32 -> bf16 convert in registers
  int brow = tid >> 2, bcol = (tid & 3) * 8;
  const float* gp = wg + ((size_t)e * I_DIM + (size_t)(n0 + brow)) * H_DIM + bcol;
  const float* up = wu + ((size_t)e * I_DIM + (size_t)(n0 + brow)) * H_DIM + bcol;

  v8f accG[4] = {};
  v8f accU[4] = {};

  // fragment indices (ISA 16-bit A layout: lanes 0-15 K {0..7,16..23}, lanes 16-31 +8)
  int am  = (wave << 4) + (lane & 15);
  int akb = (lane >> 4) << 3;       // 0 or 8
  int kk  = (lane >> 4) << 3;       // B: 0 or 8

  for (int k0 = 0; k0 < H_DIM; k0 += TK) {
    unsigned long long ga = gaA + (unsigned long long)(k0 * 2);
    ASYNC_LD16(ldsA, ga);
    ASYNC_LD16_O16(ldsA, ga);
    {
      const float4* p4 = (const float4*)(gp + k0);
      *(uint4*)&lBg[brow * KP + bcol] = pack8(p4[0], p4[1]);
      p4 = (const float4*)(up + k0);
      *(uint4*)&lBu[brow * KP + bcol] = pack8(p4[0], p4[1]);
    }
    WAIT_ASYNC0();
    __syncthreads();

    AF a;
    a.u[0] = *(const uint4*)&lA[am * KP + akb];        // K = akb .. akb+7
    a.u[1] = *(const uint4*)&lA[am * KP + akb + 16];   // K = akb+16 .. akb+23
#pragma unroll
    for (int nt = 0; nt < 4; ++nt) {
      int bn = (nt << 4) + (lane & 15);
      AF bg, bu;
      bg.u[0] = *(const uint4*)&lBg[bn * KP + kk];
      bg.u[1] = *(const uint4*)&lBg[bn * KP + kk + 16];
      bu.u[0] = *(const uint4*)&lBu[bn * KP + kk];
      bu.u[1] = *(const uint4*)&lBu[bn * KP + kk + 16];
      accG[nt] = __builtin_amdgcn_wmma_f32_16x16x32_bf16(false, a.v, false, bg.v,
                                                         (short)0, accG[nt], false, false);
      accU[nt] = __builtin_amdgcn_wmma_f32_16x16x32_bf16(false, a.v, false, bu.v,
                                                         (short)0, accU[nt], false, false);
    }
    __syncthreads();
  }

  // epilogue: exact gelu(g)*u, store bf16
  int mhalf = (lane >> 4) << 3;  // C/D layout: lanes 16-31 hold M=r+8
#pragma unroll
  for (int nt = 0; nt < 4; ++nt) {
    int col = n0 + (nt << 4) + (lane & 15);
#pragma unroll
    for (int r = 0; r < 8; ++r) {
      int m = (wave << 4) + mhalf + r;
      if (m < rowsLeft) {
        float g = accG[nt][r];
        float u = accU[nt][r];
        float hg = 0.5f * g * (1.0f + erff(g * 0.70710678118654752f)) * u;
        upbuf[(size_t)(base + mb * TM + m) * I_DIM + col] = bf16bits(hg);
      }
    }
  }
}

// ---------------- GEMM2: out += (up @ Wd^T) * w, scatter-add ----------------
__global__ __launch_bounds__(256) void k_down(const unsigned short* __restrict__ upbuf,
                                              const float* __restrict__ wd,
                                              const int* __restrict__ eoff,
                                              const int* __restrict__ perm_tok,
                                              const float* __restrict__ perm_w,
                                              float* __restrict__ out) {
  int e = blockIdx.y;
  int base = eoff[e], count = eoff[e + 1] - base;
  int mb = blockIdx.x;
  if (mb * TM >= count) return;
  int rowsLeft = count - mb * TM; if (rowsLeft > TM) rowsLeft = TM;
  int n0 = blockIdx.z * TN;

  __shared__ unsigned short lA[TM * KP];
  __shared__ unsigned short lB[TN * KP];

  int tid = threadIdx.x;
  int wave = tid >> 5, lane = tid & 31;

  // A (up activations, already bf16): async copy; clamp tail rows in-bounds,
  // garbage rows are never stored.
  int arow = tid >> 1, acol = (tid & 1) * 16;
  int grow = base + mb * TM + arow;
  if (grow >= PAIRS) grow = PAIRS - 1;
  unsigned long long gaA =
      (unsigned long long)(size_t)(upbuf + (size_t)grow * I_DIM + acol);
  unsigned ldsA = (unsigned)(size_t)&lA[arow * KP + acol];

  int brow = tid >> 2, bcol = (tid & 3) * 8;
  const float* bp = wd + ((size_t)e * H_DIM + (size_t)(n0 + brow)) * I_DIM + bcol;

  v8f acc[4] = {};

  int am  = (wave << 4) + (lane & 15);
  int akb = (lane >> 4) << 3;
  int kk  = (lane >> 4) << 3;

  for (int k0 = 0; k0 < I_DIM; k0 += TK) {
    unsigned long long ga = gaA + (unsigned long long)(k0 * 2);
    ASYNC_LD16(ldsA, ga);
    ASYNC_LD16_O16(ldsA, ga);
    {
      const float4* p4 = (const float4*)(bp + k0);
      *(uint4*)&lB[brow * KP + bcol] = pack8(p4[0], p4[1]);
    }
    WAIT_ASYNC0();
    __syncthreads();

    AF a;
    a.u[0] = *(const uint4*)&lA[am * KP + akb];
    a.u[1] = *(const uint4*)&lA[am * KP + akb + 16];
#pragma unroll
    for (int nt = 0; nt < 4; ++nt) {
      int bn = (nt << 4) + (lane & 15);
      AF b;
      b.u[0] = *(const uint4*)&lB[bn * KP + kk];
      b.u[1] = *(const uint4*)&lB[bn * KP + kk + 16];
      acc[nt] = __builtin_amdgcn_wmma_f32_16x16x32_bf16(false, a.v, false, b.v,
                                                        (short)0, acc[nt], false, false);
    }
    __syncthreads();
  }

  int mhalf = (lane >> 4) << 3;
#pragma unroll
  for (int r = 0; r < 8; ++r) {
    int m = (wave << 4) + mhalf + r;
    if (m < rowsLeft) {
      int prow = base + mb * TM + m;
      int tok = perm_tok[prow];
      float wgt = perm_w[prow];
      float* orow = out + (size_t)tok * H_DIM;
#pragma unroll
      for (int nt = 0; nt < 4; ++nt) {
        int col = n0 + (nt << 4) + (lane & 15);
        unsafeAtomicAdd(&orow[col], acc[nt][r] * wgt);
      }
    }
  }
}

// ---------------- Host launcher ---------------------------------------------
extern "C" void kernel_launch(void* const* d_in, const int* in_sizes, int n_in,
                              void* d_out, int out_size, void* d_ws, size_t ws_size,
                              hipStream_t stream) {
  (void)in_sizes; (void)n_in; (void)out_size; (void)ws_size;
  const float* hidden = (const float*)d_in[0];
  const float* gate_w = (const float*)d_in[1];
  const float* w_gate = (const float*)d_in[2];
  const float* w_up   = (const float*)d_in[3];
  const float* w_down = (const float*)d_in[4];
  float* out = (float*)d_out;                         // expert_out: T x H
  float* logits = out + (size_t)T_TOK * H_DIM;        // router_logits: T x E

  // workspace layout
  int*   i1 = (int*)d_ws;
  int*   i2 = i1 + T_TOK;
  float* w1 = (float*)(i2 + T_TOK);
  float* w2 = w1 + T_TOK;
  int*   eoff = (int*)(w2 + T_TOK);                   // 9 used
  int*   perm_tok = eoff + 16;                        // PAIRS
  float* perm_w = (float*)(perm_tok + PAIRS);         // PAIRS
  unsigned short* upbuf = (unsigned short*)((char*)d_ws + (1 << 19));       // 128 MB
  unsigned short* xbf = upbuf + (size_t)PAIRS * I_DIM;                      // +16 MB

  hipMemsetAsync(out, 0, (size_t)T_TOK * H_DIM * sizeof(float), stream);
  k_cvtx<<<(T_TOK * (H_DIM / 8)) / 256, 256, 0, stream>>>(hidden, xbf);
  k_router<<<T_TOK / 8, 256, 0, stream>>>(hidden, gate_w, logits);
  k_top2<<<T_TOK / 256, 256, 0, stream>>>(logits, i1, i2, w1, w2);
  k_build<<<1, 256, 0, stream>>>(i1, i2, w1, w2, eoff, perm_tok, perm_w);
  k_up<<<dim3(32, N_EXP, I_DIM / TN), 256, 0, stream>>>(xbf, w_gate, w_up, eoff, perm_tok, upbuf);
  k_down<<<dim3(32, N_EXP, H_DIM / TN), 256, 0, stream>>>(upbuf, w_down, eoff, perm_tok, perm_w, out);
}
